// ProgressivePruningSystem_28587302322862
// MI455X (gfx1250) — matrix operand order repaired and checked by reading, not compiled, not run on hardware
//
#include <hip/hip_runtime.h>
#include <hip/hip_fp16.h>
#include <stdint.h>

// ---------------------------------------------------------------------------
// Problem constants (from the reference)
// ---------------------------------------------------------------------------
#define NTOK   16384       // B*L
#define DDIM   1024        // D
#define FFDIM  2048        // FF
#define NHEAD  16          // H
#define NPATH  5           // P
#define DH     64          // D/H
#define HPDIM  80          // H*P
#define ENTROPY_COEFF 0.02f
#define FLOOR_MAX     0.05f

typedef _Float16 v16h __attribute__((ext_vector_type(16)));
typedef _Float16 v8h  __attribute__((ext_vector_type(8)));
typedef _Float16 v4h  __attribute__((ext_vector_type(4)));
typedef float    v8f  __attribute__((ext_vector_type(8)));

// ---------------------------------------------------------------------------
// Prep kernels (pure bandwidth -> vectorized b128 loads)
// ---------------------------------------------------------------------------
__global__ void cvt_f32_to_f16(const float* __restrict__ src,
                               _Float16* __restrict__ dst, int n4) {
    int stride = gridDim.x * blockDim.x;
    for (int i = blockIdx.x * blockDim.x + threadIdx.x; i < n4; i += stride) {
        float4 v = ((const float4*)src)[i];
        v4h h; h.x = (_Float16)v.x; h.y = (_Float16)v.y;
               h.z = (_Float16)v.z; h.w = (_Float16)v.w;
        ((v4h*)dst)[i] = h;
    }
}

__global__ void dequant_i8_f16(const int8_t* __restrict__ q,
                               const float* __restrict__ scale,
                               _Float16* __restrict__ dst, int n4, int k) {
    // k (row length) is a multiple of 4 -> scale constant within a packed dword
    int stride = gridDim.x * blockDim.x;
    for (int i = blockIdx.x * blockDim.x + threadIdx.x; i < n4; i += stride) {
        int packed = ((const int*)q)[i];
        float s = scale[(i * 4) / k];
        v4h h;
        h.x = (_Float16)((float)(int8_t)(packed)       * s);
        h.y = (_Float16)((float)(int8_t)(packed >> 8)  * s);
        h.z = (_Float16)((float)(int8_t)(packed >> 16) * s);
        h.w = (_Float16)((float)(int8_t)(packed >> 24) * s);
        ((v4h*)dst)[i] = h;
    }
}

// ---------------------------------------------------------------------------
// GEMM1 + exact-erf GELU:  Hh[m,n] = gelu( A[m,:] . W1[n,:] + b1[n] )
//   A  : (NTOK, DDIM)  f16 row-major
//   W1 : (FFDIM, DDIM) f16 row-major (N-major, K contiguous -> ideal B frag)
// Wave tile 64(M) x 32(N): acc[4][2], 4 A-frags + 2 B-frags per K-step,
// 8 WMMAs per 6 fragment loads.  Block = 8 waves = 128(M) x 128(N).
// ---------------------------------------------------------------------------
__global__ __launch_bounds__(256)
void gemm1_gelu(const _Float16* __restrict__ A,
                const _Float16* __restrict__ W,
                const float* __restrict__ b1,
                _Float16* __restrict__ Hh) {
    const int lane = threadIdx.x & 31;
    const int wid  = threadIdx.x >> 5;
    const int lr   = lane & 15;       // row/col within 16
    const int hi   = lane >> 4;       // lane half selects K split
    const int m0   = blockIdx.y * 128 + (wid >> 2) * 64;   // 2 wave rows
    const int n0   = blockIdx.x * 128 + (wid & 3) * 32;    // 4 wave cols

    v8f acc[4][2] = {};
    for (int k0 = 0; k0 < DDIM; k0 += 32) {
        v16h af[4], bf[2];
#pragma unroll
        for (int i = 0; i < 4; ++i) {
            // A fragment (16x32 f16): lane<16 -> K {0..15},
            // lane>=16 -> K {16..31}, M = lr (matches B fragment K order).
            const _Float16* ap = A + (size_t)(m0 + i * 16 + lr) * DDIM + k0 + hi * 16;
            union { v16h v; v8h h[2]; } u;
            u.h[0] = *(const v8h*)(ap);
            u.h[1] = *(const v8h*)(ap + 8);
            af[i] = u.v;
        }
#pragma unroll
        for (int j = 0; j < 2; ++j) {
            // B fragment (32x16 f16): lanes 0-15 hold K=0..15, lanes 16-31
            // hold K=16..31, N = lr; contiguous 32B per lane in (N,K)-major W1.
            bf[j] = *(const v16h*)(W + (size_t)(n0 + j * 16 + lr) * DDIM + k0 + hi * 16);
        }
#pragma unroll
        for (int i = 0; i < 4; ++i)
#pragma unroll
            for (int j = 0; j < 2; ++j)
                acc[i][j] = __builtin_amdgcn_wmma_f32_16x16x32_f16(
                    false, af[i], false, bf[j], (short)0, acc[i][j], false, false);
    }

    // Epilogue: +bias, exact GELU, store f16
#pragma unroll
    for (int i = 0; i < 4; ++i)
#pragma unroll
        for (int j = 0; j < 2; ++j) {
            const int col = n0 + j * 16 + lr;
            const float bb = b1[col];
#pragma unroll
            for (int r = 0; r < 8; ++r) {
                const int row = m0 + i * 16 + r + hi * 8;   // C layout: VGPR r -> M=r (+8 upper lanes)
                float x = acc[i][j][r] + bb;
                float g = 0.5f * x * (1.0f + erff(x * 0.70710678118654752f));
                Hh[(size_t)row * FFDIM + col] = (_Float16)g;
            }
        }
}

// ---------------------------------------------------------------------------
// GEMM2 (logits) + temperature softmax + entropy + eps-floor + path combine.
// Block: 256 threads = 8 waves, 128 tokens per block (16 tokens per wave).
// ---------------------------------------------------------------------------
__global__ __launch_bounds__(256)
void gate_combine(const _Float16* __restrict__ Hh,
                  const _Float16* __restrict__ W2h,   // (HPDIM, FFDIM) f16
                  const float* __restrict__ b2,        // (HPDIM)
                  const float* __restrict__ gate_log_temp,   // (NHEAD)
                  const float* __restrict__ gate_eps_logit,  // (NHEAD*NPATH)
                  const float* __restrict__ p0, const float* __restrict__ p1,
                  const float* __restrict__ p2, const float* __restrict__ p3,
                  const float* __restrict__ p4,
                  float* __restrict__ out,
                  float* __restrict__ ent_partial) {
    __shared__ float lds[128 * HPDIM];   // logits, then floored probs (in place)
    __shared__ float red[256];

    const int lane = threadIdx.x & 31;
    const int wid  = threadIdx.x >> 5;
    const int lr   = lane & 15;
    const int hi   = lane >> 4;
    const int tb   = blockIdx.x * 128;

    // ---- GEMM2: 16 tokens x 80 logits per wave, K = 2048 ----
    {
        const int trow = tb + wid * 16;
        v8f acc[5] = {};
        for (int k0 = 0; k0 < FFDIM; k0 += 32) {
            const _Float16* ap = Hh + (size_t)(trow + lr) * FFDIM + k0 + hi * 16;
            union { v16h v; v8h h[2]; } u;
            u.h[0] = *(const v8h*)(ap);
            u.h[1] = *(const v8h*)(ap + 8);
            const v16h a = u.v;
#pragma unroll
            for (int t = 0; t < 5; ++t) {
                v16h b = *(const v16h*)(W2h + (size_t)(t * 16 + lr) * FFDIM + k0 + hi * 16);
                acc[t] = __builtin_amdgcn_wmma_f32_16x16x32_f16(
                    false, a, false, b, (short)0, acc[t], false, false);
            }
        }
#pragma unroll
        for (int t = 0; t < 5; ++t)
#pragma unroll
            for (int r = 0; r < 8; ++r) {
                const int ltok = wid * 16 + r + hi * 8;
                const int col  = t * 16 + lr;
                lds[ltok * HPDIM + col] = acc[t][r] + b2[col];
            }
    }
    __syncthreads();

    // ---- per-(token,head): temp scale, softmax, entropy, eps floor ----
    float esum = 0.0f;
    for (int pair = threadIdx.x; pair < 128 * NHEAD; pair += 256) {
        const int tok = pair >> 4;
        const int hh  = pair & 15;
        float* lg = &lds[tok * HPDIM + hh * NPATH];
        const float temp = fmaxf(log1pf(expf(gate_log_temp[hh])), 1e-4f); // softplus
        float l[NPATH], mx = -1e30f;
#pragma unroll
        for (int p = 0; p < NPATH; ++p) { l[p] = lg[p] / temp; mx = fmaxf(mx, l[p]); }
        float s = 0.0f;
#pragma unroll
        for (int p = 0; p < NPATH; ++p) { l[p] = expf(l[p] - mx); s += l[p]; }
        const float inv = 1.0f / s;
        float pmax = 0.0f;
#pragma unroll
        for (int p = 0; p < NPATH; ++p) { l[p] *= inv; pmax = fmaxf(pmax, l[p]); }
        // entropy from pre-floor probs
#pragma unroll
        for (int p = 0; p < NPATH; ++p) {
            const float cp = fminf(fmaxf(l[p], 1e-9f), 1.0f);
            esum -= l[p] * logf(cp);
        }
        // token-adaptive epsilon floor
        float eps[NPATH], epss = 0.0f;
#pragma unroll
        for (int p = 0; p < NPATH; ++p) {
            const float eb = 1.0f / (1.0f + expf(-gate_eps_logit[hh * NPATH + p]));
            eps[p] = FLOOR_MAX * (1.0f - pmax) * eb;
            epss += eps[p];
        }
#pragma unroll
        for (int p = 0; p < NPATH; ++p) {
            const float v = l[p] * (1.0f - epss) + eps[p];
            lg[p] = fminf(fmaxf(v, 1e-9f), 1.0f);
        }
    }

    // ---- deterministic block entropy reduction ----
    red[threadIdx.x] = esum;
    __syncthreads();
    for (int off = 128; off > 0; off >>= 1) {
        if (threadIdx.x < off) red[threadIdx.x] += red[threadIdx.x + off];
        __syncthreads();
    }
    if (threadIdx.x == 0) ent_partial[blockIdx.x] = red[0];

    // ---- weighted path combine: float4 per thread, one token per sweep ----
    {
        const int e  = threadIdx.x * 4;       // 0..1020, covers D=1024 exactly
        const int hh = e >> 6;                // head constant within float4
        for (int t = 0; t < 128; ++t) {
            const size_t gi = ((size_t)(tb + t)) * (NHEAD * DH) + e;
            const float* w = &lds[t * HPDIM + hh * NPATH];
            if (t + 1 < 128) {   // prefetch next token's path lines (global_prefetch_b8)
                __builtin_prefetch(p0 + gi + NHEAD * DH, 0, 0);
                __builtin_prefetch(p1 + gi + NHEAD * DH, 0, 0);
                __builtin_prefetch(p2 + gi + NHEAD * DH, 0, 0);
                __builtin_prefetch(p3 + gi + NHEAD * DH, 0, 0);
                __builtin_prefetch(p4 + gi + NHEAD * DH, 0, 0);
            }
            const float4 a0 = *(const float4*)(p0 + gi);
            const float4 a1 = *(const float4*)(p1 + gi);
            const float4 a2 = *(const float4*)(p2 + gi);
            const float4 a3 = *(const float4*)(p3 + gi);
            const float4 a4 = *(const float4*)(p4 + gi);
            float4 r;
            r.x = w[0]*a0.x + w[1]*a1.x + w[2]*a2.x + w[3]*a3.x + w[4]*a4.x;
            r.y = w[0]*a0.y + w[1]*a1.y + w[2]*a2.y + w[3]*a3.y + w[4]*a4.y;
            r.z = w[0]*a0.z + w[1]*a1.z + w[2]*a2.z + w[3]*a3.z + w[4]*a4.z;
            r.w = w[0]*a0.w + w[1]*a1.w + w[2]*a2.w + w[3]*a3.w + w[4]*a4.w;
            *(float4*)(out + gi) = r;
        }
    }
}

// ---------------------------------------------------------------------------
// Deterministic finalize of the entropy scalar
// ---------------------------------------------------------------------------
__global__ void ent_finalize(const float* __restrict__ partials, int n,
                             float* __restrict__ dst) {
    if (blockIdx.x == 0 && threadIdx.x == 0) {
        float s = 0.0f;
        for (int i = 0; i < n; ++i) s += partials[i];
        *dst = ENTROPY_COEFF * (s / (float)(NTOK * NHEAD));
    }
}

// ---------------------------------------------------------------------------
// Launcher
// ---------------------------------------------------------------------------
extern "C" void kernel_launch(void* const* d_in, const int* in_sizes, int n_in,
                              void* d_out, int out_size, void* d_ws, size_t ws_size,
                              hipStream_t stream) {
    const float*  hidden = (const float*)d_in[0];
    const float*  p0     = (const float*)d_in[1];
    const float*  p1     = (const float*)d_in[2];
    const float*  p2     = (const float*)d_in[3];
    const float*  p3     = (const float*)d_in[4];
    const float*  p4     = (const float*)d_in[5];
    const int8_t* w1q    = (const int8_t*)d_in[6];
    const float*  s1     = (const float*)d_in[7];
    const float*  b1     = (const float*)d_in[8];
    const int8_t* w2q    = (const int8_t*)d_in[9];
    const float*  s2     = (const float*)d_in[10];
    const float*  b2     = (const float*)d_in[11];
    const float*  glt    = (const float*)d_in[12];
    const float*  gel    = (const float*)d_in[13];
    float* out = (float*)d_out;

    // Workspace layout (bytes):
    char* ws = (char*)d_ws;
    _Float16* Ah   = (_Float16*)(ws + 0);          //  33,554,432 : hidden f16
    _Float16* W1h  = (_Float16*)(ws + 33554432);   //   4,194,304 : w1 f16
    _Float16* W2h  = (_Float16*)(ws + 37748736);   //     327,680 : w2 f16
    _Float16* Hh   = (_Float16*)(ws + 38076416);   //  67,108,864 : gelu(h) f16
    float*    part = (float*)   (ws + 105185280);  //         512 : entropy partials

    cvt_f32_to_f16<<<2048, 256, 0, stream>>>(hidden, Ah, (NTOK * DDIM) / 4);
    dequant_i8_f16<<<512,  256, 0, stream>>>(w1q, s1, W1h, (FFDIM * DDIM) / 4, DDIM);
    dequant_i8_f16<<<40,   256, 0, stream>>>(w2q, s2, W2h, (HPDIM * FFDIM) / 4, FFDIM);

    dim3 g1(FFDIM / 128, NTOK / 128);
    gemm1_gelu<<<g1, 256, 0, stream>>>(Ah, W1h, b1, Hh);

    gate_combine<<<NTOK / 128, 256, 0, stream>>>(Hh, W2h, b2, glt, gel,
                                                 p0, p1, p2, p3, p4,
                                                 out, part);

    ent_finalize<<<1, 32, 0, stream>>>(part, NTOK / 128,
                                       out + (size_t)NTOK * NHEAD * DH);
}
